// BayesianGCN_37520834297969
// MI455X (gfx1250) — matrix operand order repaired and checked by reading, not compiled, not run on hardware
//
#include <hip/hip_runtime.h>

typedef __bf16 bf16_t;
typedef __attribute__((ext_vector_type(16))) __bf16 v16bf;
typedef __attribute__((ext_vector_type(8)))  float  v8f;

#define N_NODES 50000
#define N_EDGES 800000
#define MTILES  (N_NODES / 16)          /* 3125 */
#define MGROUPS ((MTILES + 7) / 8)      /* 391  */

union V16U { v16bf v; uint4 q[2]; };
union Q8U  { uint4 q; bf16_t h[8]; };

// ---------------- weight sampling: W = mu + softplus(rho) * eps ----------------

__device__ __forceinline__ unsigned hash_u32(unsigned x) {
  x ^= x >> 16; x *= 0x7feb352dU;
  x ^= x >> 15; x *= 0x846ca68bU;
  x ^= x >> 16; return x;
}

__device__ __forceinline__ float gauss(unsigned tag, unsigned idx) {
  unsigned h1 = hash_u32(tag * 0x9E3779B9U + idx * 2u + 1u);
  unsigned h2 = hash_u32(tag * 0x85EBCA6BU + idx * 2u);
  float u1 = (float)((h1 >> 8) + 1u) * (1.0f / 16777216.0f);   // (0,1]
  float u2 = (float)(h2 >> 8) * (1.0f / 16777216.0f);
  return sqrtf(-2.0f * __logf(u1)) * __cosf(6.28318530718f * u2);
}

// Writes W transposed (din x dout) in bf16 so WMMA B-fragments are contiguous,
// and bias in f32.
__global__ void sample_weights_kernel(const float* __restrict__ wmu,
                                      const float* __restrict__ wrho,
                                      const float* __restrict__ bmu,
                                      const float* __restrict__ brho,
                                      bf16_t* __restrict__ Wt,
                                      float* __restrict__ bias,
                                      int dout, int din, unsigned tag) {
  int idx = blockIdx.x * blockDim.x + threadIdx.x;
  int nW = dout * din;
  if (idx < nW) {
    int o = idx / din, i = idx - o * din;
    float sp = log1pf(__expf(wrho[idx]));
    float w  = wmu[idx] + sp * gauss(tag, (unsigned)idx);
    Wt[(size_t)i * dout + o] = (bf16_t)w;
  } else if (idx < nW + dout) {
    int o = idx - nW;
    float sp = log1pf(__expf(brho[o]));
    bias[o] = bmu[o] + sp * gauss(tag ^ 0x5bd1e995U, (unsigned)o);
  }
}

// ---------------- elementwise helpers ----------------

__global__ void f32_to_bf16_kernel(const float* __restrict__ in,
                                   bf16_t* __restrict__ out, int n) {
  int i = blockIdx.x * blockDim.x + threadIdx.x;
  if (i < n) out[i] = (bf16_t)in[i];
}

__global__ void fill_zero_kernel(float* __restrict__ p, int n) {
  int i = blockIdx.x * blockDim.x + threadIdx.x;
  if (i < n) p[i] = 0.0f;
}

__global__ void relu_f32_to_bf16_kernel(const float* __restrict__ in,
                                        bf16_t* __restrict__ out, int n) {
  int i = blockIdx.x * blockDim.x + threadIdx.x;
  if (i < n) { float v = in[i]; out[i] = (bf16_t)(v > 0.0f ? v : 0.0f); }
}

// ---------------- WMMA GEMM with TDM/LDS-staged weights -----------------------
// Block = one 16-col tile x 8 consecutive 16-row tiles (8 waves).
// The din x 16 weight slab is DMA'd into LDS once per block via the Tensor
// Data Mover (fallback: cooperative b128 copy), then each wave's K-loop reads
// B fragments from LDS (strided 16-bit reads) and A fragments from global.
//
// A-fragment layout (16-bit 16x32): lanes 0-15 row M=lane, K {k0+0..7, k0+16..23};
// lanes 16-31 row M=lane-16, K {k0+8..15, k0+24..31}.
// B-fragment: col N=lane&15, K same as A (LDS stride 16 per K).

#if defined(__has_builtin)
#if __has_builtin(__builtin_amdgcn_tensor_load_to_lds) && \
    __has_builtin(__builtin_amdgcn_s_wait_tensorcnt)
#define USE_TDM 1
#endif
#endif
#ifndef USE_TDM
#define USE_TDM 0
#endif

#if USE_TDM
typedef __attribute__((ext_vector_type(4))) unsigned tdm_g0_t;
typedef __attribute__((ext_vector_type(8))) int      tdm_g1_t;
typedef __attribute__((ext_vector_type(4))) int      tdm_g2_t;
typedef __attribute__((ext_vector_type(8))) int      tdm_g4_t;
#endif

template <bool RELU>
__global__ void gemm_wmma_lds_kernel(const bf16_t* __restrict__ A,
                                     const bf16_t* __restrict__ Bt,  // [din,dout]
                                     const float* __restrict__ bias,
                                     bf16_t* __restrict__ Out,
                                     int mtiles, int din, int dout, int tiles_n) {
  __shared__ bf16_t Bs[128 * 16];          // din_max x 16 slab, 4 KB

  int tn = blockIdx.x % tiles_n;
  int mgroup = blockIdx.x / tiles_n;
  int n0 = tn * 16;

#if USE_TDM
  if (threadIdx.x < 32) {                  // wave 0 issues one TDM descriptor
    unsigned long long gaddr = (unsigned long long)(const void*)(Bt + n0);
    tdm_g0_t g0;
    g0[0] = 1u;                                          // count=1, user D#
    g0[1] = 0u;                                          // lds_addr: Bs at LDS 0
    g0[2] = (unsigned)gaddr;                             // global_addr[31:0]
    g0[3] = ((unsigned)(gaddr >> 32) & 0x01FFFFFFu)      // global_addr[56:32]
            | 0x80000000u;                               // type=2 ("image")
    tdm_g1_t g1;
    g1[0] = (int)0x00010000u;                            // data_size=2B, no mask
    g1[1] = (int)(((unsigned)dout & 0xFFFFu) << 16);     // tensor_dim0[15:0]
    g1[2] = (int)(((unsigned)din & 0xFFFFu) << 16);      // dim0 hi=0 | tensor_dim1 lo
    g1[3] = (int)(16u << 16);                            // dim1 hi=0 | tile_dim0=16
    g1[4] = din;                                         // tile_dim1=din, tile_dim2=0
    g1[5] = dout;                                        // tensor_dim0_stride lo
    g1[6] = 0;                                           // stride hi | dim1_stride lo
    g1[7] = 0;
    tdm_g2_t z4 = {0, 0, 0, 0};
    tdm_g4_t z8 = {0, 0, 0, 0, 0, 0, 0, 0};
    __builtin_amdgcn_tensor_load_to_lds(g0, g1, z4, z4, z8, 0);
    __builtin_amdgcn_s_wait_tensorcnt(0);
  }
#else
  for (int t = threadIdx.x; t < din * 2; t += blockDim.x) {
    int k = t >> 1, p = t & 1;
    *(uint4*)(Bs + k * 16 + p * 8) =
        *(const uint4*)(Bt + (size_t)k * dout + n0 + p * 8);
  }
#endif
  __syncthreads();

  int wave = threadIdx.x >> 5;
  int lane = threadIdx.x & 31;
  int mt = mgroup * 8 + wave;
  if (mt >= mtiles) return;                // wave-uniform: EXEC all-1 at WMMA
  int m0 = mt << 4;

  int half = lane >> 4;
  int l16  = lane & 15;
  const bf16_t* arow = A + (size_t)(m0 + l16) * din + half * 8;
  const bf16_t* bls  = Bs + half * 8 * 16 + l16;  // col N=l16, K from half*8; stride 16 per K

  v8f acc = {};
  for (int k0 = 0; k0 < din; k0 += 32) {
    V16U a, b;
    a.q[0] = *(const uint4*)(arow + k0);           // K = k0 + half*8 + 0..7
    a.q[1] = *(const uint4*)(arow + k0 + 16);      // K = k0 + half*8 + 16..23
#pragma unroll
    for (int s = 0; s < 8; ++s) {                  // K = k0 + half*8 + {0..7, 16..23}
      b.v[s]     = bls[(k0 + s) * 16];
      b.v[8 + s] = bls[(k0 + 16 + s) * 16];
    }
    acc = __builtin_amdgcn_wmma_f32_16x16x32_bf16(
        /*neg_a=*/false, a.v, /*neg_b=*/false, b.v,
        /*c_mod=*/(short)0, acc, /*reuse_a=*/false, /*reuse_b=*/false);
  }

  // C/D layout: col = n0 + (lane&15); VGPR r -> row m0 + r + 8*(lane>=16)
  int col = n0 + l16;
  float bv = bias[col];
#pragma unroll
  for (int r = 0; r < 8; ++r) {
    int row = m0 + r + half * 8;
    float v = acc[r] + bv;
    if (RELU) v = v > 0.0f ? v : 0.0f;
    Out[(size_t)row * dout + col] = (bf16_t)v;
  }
}

// ---------------- SpMM: S[dst] += w_e * T[src]  (bf16 gather, f32 atomic scatter)

__global__ void spmm_bf16_kernel(const int* __restrict__ ei,   // [2,E] flat
                                 const float* __restrict__ ew, // [E]
                                 const bf16_t* __restrict__ T, // [N,F] bf16
                                 float* __restrict__ S,        // [N,F] f32 (zeroed)
                                 int nedges, int F) {
  int t = blockIdx.x * blockDim.x + threadIdx.x;
  int cpr = F >> 3;                          // 8-feature chunks per edge
  int total = nedges * cpr;
  if (t >= total) return;
  int e = t / cpr;
  int c = t - e * cpr;
  int s = ei[e];
  int d = ei[nedges + e];
  float w = ew[e];
  Q8U u;
  u.q = *(const uint4*)(T + (size_t)s * F + c * 8);
  float* outp = S + (size_t)d * F + c * 8;
#pragma unroll
  for (int j = 0; j < 8; ++j) atomicAdd(outp + j, w * (float)u.h[j]);
}

// ---------------- final 32 -> 2 layer (dout below WMMA tile width) -------------

__global__ void final_layer_kernel(const bf16_t* __restrict__ h,
                                   const bf16_t* __restrict__ Wt,  // [din,dout]
                                   const float* __restrict__ bias,
                                   float* __restrict__ out,
                                   int nrows, int din, int dout) {
  int i = blockIdx.x * blockDim.x + threadIdx.x;
  if (i >= nrows) return;
  for (int o = 0; o < dout; ++o) {
    float s = bias[o];
    for (int k = 0; k < din; ++k)
      s += (float)h[(size_t)i * din + k] * (float)Wt[(size_t)k * dout + o];
    out[(size_t)i * dout + o] = s;
  }
}

// ---------------- orchestration ----------------

extern "C" void kernel_launch(void* const* d_in, const int* in_sizes, int n_in,
                              void* d_out, int out_size, void* d_ws, size_t ws_size,
                              hipStream_t stream) {
  const float* x  = (const float*)d_in[0];
  const int*   ei = (const int*)d_in[1];
  const float* ew = (const float*)d_in[2];

  const int DIN[6]  = {128, 128, 128, 64, 64, 32};
  const int DOUT[6] = {128, 128,  64, 64, 32,  2};

  char* ws = (char*)d_ws;
  size_t off = 0;
  auto take = [&](size_t bytes) {
    size_t o = off;
    off += (bytes + 255) & ~(size_t)255;
    return o;
  };

  bf16_t* Wt[6];
  float*  bias[6];
  for (int l = 0; l < 6; ++l) {
    Wt[l]   = (bf16_t*)(ws + take((size_t)DIN[l] * DOUT[l] * sizeof(bf16_t)));
    bias[l] = (float*)(ws + take((size_t)DOUT[l] * sizeof(float)));
  }
  bf16_t* hA = (bf16_t*)(ws + take((size_t)N_NODES * 128 * sizeof(bf16_t)));
  bf16_t* hB = (bf16_t*)(ws + take((size_t)N_NODES * 128 * sizeof(bf16_t)));
  float*  S  = (float*)(ws + take((size_t)N_NODES * 128 * sizeof(float)));

  // 1) sample Bayesian weights (tiny)
  for (int l = 0; l < 6; ++l) {
    const float* wmu  = (const float*)d_in[3 + 4 * l + 0];
    const float* wrho = (const float*)d_in[3 + 4 * l + 1];
    const float* bmu  = (const float*)d_in[3 + 4 * l + 2];
    const float* brho = (const float*)d_in[3 + 4 * l + 3];
    int n = DOUT[l] * DIN[l] + DOUT[l];
    sample_weights_kernel<<<(n + 255) / 256, 256, 0, stream>>>(
        wmu, wrho, bmu, brho, Wt[l], bias[l], DOUT[l], DIN[l], 42u + (unsigned)l);
  }

  // 2) x -> bf16
  {
    int n = N_NODES * 128;
    f32_to_bf16_kernel<<<(n + 255) / 256, 256, 0, stream>>>(x, hA, n);
  }

  bf16_t* cur = hA;
  bf16_t* tmp = hB;

  // 3) three GCN layers: gemm -> spmm(atomic) -> relu
  for (int l = 0; l < 3; ++l) {
    int din = DIN[l], dout = DOUT[l];
    int tiles_n = dout / 16;
    gemm_wmma_lds_kernel<false><<<MGROUPS * tiles_n, 256, 0, stream>>>(
        cur, Wt[l], bias[l], tmp, MTILES, din, dout, tiles_n);
    int n = N_NODES * dout;
    fill_zero_kernel<<<(n + 255) / 256, 256, 0, stream>>>(S, n);
    int chunks = N_EDGES * (dout / 8);
    spmm_bf16_kernel<<<(chunks + 255) / 256, 256, 0, stream>>>(ei, ew, tmp, S,
                                                               N_EDGES, dout);
    relu_f32_to_bf16_kernel<<<(n + 255) / 256, 256, 0, stream>>>(S, cur, n);
  }

  // 4) two MLP layers with fused bias+relu
  for (int l = 3; l < 5; ++l) {
    int din = DIN[l], dout = DOUT[l];
    int tiles_n = dout / 16;
    gemm_wmma_lds_kernel<true><<<MGROUPS * tiles_n, 256, 0, stream>>>(
        cur, Wt[l], bias[l], tmp, MTILES, din, dout, tiles_n);
    bf16_t* t = cur; cur = tmp; tmp = t;
  }

  // 5) final 32 -> 2 layer, f32 output, no relu
  final_layer_kernel<<<(N_NODES + 255) / 256, 256, 0, stream>>>(
      cur, Wt[5], bias[5], (float*)d_out, N_NODES, 32, 2);
}
